// RGATtest_4105988735705
// MI455X (gfx1250) — compile-verified, matrix operand order checked by reading, not run on hardware
//
#include <hip/hip_runtime.h>
#include <math.h>

#define N_NODES 50000
#define N_EDGES 500000
#define NHEAD 4
#define DHEAD 32
#define FEAT 128
#define BN_EPS 1e-5f
#define NEG_SLOPE 0.2f

typedef __attribute__((ext_vector_type(16))) __bf16 v16bf;
typedef __attribute__((ext_vector_type(8)))  float  v8f;

// ---------------------------------------------------------------------------
// Generic fill
// ---------------------------------------------------------------------------
__global__ void rg_fill(float* __restrict__ p, float v, long n) {
    long i = (long)blockIdx.x * blockDim.x + threadIdx.x;
    if (i < n) p[i] = v;
}

// ---------------------------------------------------------------------------
// WMMA bf16 GEMM:  Y[M,Nc] = A[M,K] @ B[K,Nc] (+ bias[Nc])
// One wave computes a 16 x (16*TN) output strip; A fragment is reused across
// the TN column tiles. K multiple of 32, M multiple of 16, Nc multiple of 16*TN.
// A-fragment layout (16-bit A 16x32, ISA 7.12.2): lane L holds row L&15;
//   vector elem i -> k = (i>>3)*16 + (L>=16 ? 8 : 0) + (i&7)   (runs of 8
//   contiguous k -> 4 x float4 loads per k-step).
// B-fragment: symmetric (col = lane&15, same k mapping).
// C/D: VGPR j -> row j + (lane>=16 ? 8 : 0), col = lane&15.
// ---------------------------------------------------------------------------
template <int TN>
__global__ void rg_wmma_gemm(const float* __restrict__ A,
                             const float* __restrict__ B,
                             const float* __restrict__ bias,
                             float* __restrict__ Y,
                             int M, int K, int Nc) {
    const int lane = threadIdx.x & 31;
    const int wave = threadIdx.x >> 5;
    const int colGroups = Nc / (16 * TN);
    const int nTiles = (M >> 4) * colGroups;
    const int tile = blockIdx.x * (blockDim.x >> 5) + wave;
    if (tile >= nTiles) return;                 // wave-uniform exit (EXEC all-1 for WMMA)
    const int rowBase = (tile / colGroups) << 4;
    const int colBase = (tile % colGroups) * (16 * TN);
    const bool hi = lane >= 16;
    const int lr = lane & 15;
    const int koff = hi ? 8 : 0;

    // hoist bias: invariant per lane/column tile (no branch in epilogue)
    float bv[TN];
#pragma unroll
    for (int t = 0; t < TN; ++t)
        bv[t] = bias ? bias[colBase + t * 16 + lr] : 0.f;

    v8f acc[TN];
#pragma unroll
    for (int t = 0; t < TN; ++t) acc[t] = (v8f){};

    const float* __restrict__ arow = A + (long)(rowBase + lr) * K;

    for (int k0 = 0; k0 < K; k0 += 32) {
        // A fragment: 16 floats as 4 x b128 loads, then pack to bf16
        const float4 a0 = *(const float4*)(arow + k0 + koff);
        const float4 a1 = *(const float4*)(arow + k0 + koff + 4);
        const float4 a2 = *(const float4*)(arow + k0 + 16 + koff);
        const float4 a3 = *(const float4*)(arow + k0 + 16 + koff + 4);
        v16bf a;
        a[0]  = (__bf16)a0.x; a[1]  = (__bf16)a0.y; a[2]  = (__bf16)a0.z; a[3]  = (__bf16)a0.w;
        a[4]  = (__bf16)a1.x; a[5]  = (__bf16)a1.y; a[6]  = (__bf16)a1.z; a[7]  = (__bf16)a1.w;
        a[8]  = (__bf16)a2.x; a[9]  = (__bf16)a2.y; a[10] = (__bf16)a2.z; a[11] = (__bf16)a2.w;
        a[12] = (__bf16)a3.x; a[13] = (__bf16)a3.y; a[14] = (__bf16)a3.z; a[15] = (__bf16)a3.w;

#pragma unroll
        for (int t = 0; t < TN; ++t) {
            v16bf b;
#pragma unroll
            for (int i = 0; i < 16; ++i) {
                const int k = k0 + ((i >> 3) << 4) + koff + (i & 7);
                b[i] = (__bf16)B[(long)k * Nc + colBase + t * 16 + lr];
            }
            acc[t] = __builtin_amdgcn_wmma_f32_16x16x32_bf16(
                /*neg_a=*/false, a, /*neg_b=*/false, b,
                /*c_mod=*/(short)0, acc[t], /*reuse_a=*/false, /*reuse_b=*/false);
        }
    }

#pragma unroll
    for (int t = 0; t < TN; ++t) {
#pragma unroll
        for (int j = 0; j < 8; ++j) {
            const int row = rowBase + j + (hi ? 8 : 0);
            Y[(long)row * Nc + colBase + t * 16 + lr] = acc[t][j] + bv[t];
        }
    }
}

// ---------------------------------------------------------------------------
// el[n,h] = sum_d fs[n,h,d]*attn_l[h,d] ; er analogous
// ---------------------------------------------------------------------------
__global__ void rg_node_attn(const float* __restrict__ fs,
                             const float* __restrict__ al,
                             const float* __restrict__ ar,
                             float* __restrict__ el, float* __restrict__ er, int n) {
    int i = blockIdx.x * blockDim.x + threadIdx.x;   // over n*NHEAD
    if (i >= n * NHEAD) return;
    const int node = i >> 2, h = i & 3;
    const float* f = fs + (long)node * FEAT + h * DHEAD;
    const float* wl = al + h * DHEAD;
    const float* wr = ar + h * DHEAD;
    float sl = 0.f, sr = 0.f;
#pragma unroll
    for (int d = 0; d < DHEAD; ++d) { sl += f[d] * wl[d]; sr += f[d] * wr[d]; }
    el[i] = sl; er[i] = sr;
}

__device__ __forceinline__ void rg_atomic_max_f32(float* addr, float value) {
    unsigned* ua = (unsigned*)addr;
    unsigned old = *ua;
    while (__uint_as_float(old) < value) {
        unsigned assumed = old;
        old = atomicCAS(ua, assumed, __float_as_uint(value));
        if (old == assumed) break;
    }
}

// pass 1: e = leaky_relu(el[src]+er[dst]); segment max into m[dst]
__global__ void rg_edge_logits_max(const float* __restrict__ el,
                                   const float* __restrict__ er,
                                   const int* __restrict__ src,
                                   const int* __restrict__ dst,
                                   float* __restrict__ e_out,
                                   float* __restrict__ m, int E) {
    int i = blockIdx.x * blockDim.x + threadIdx.x;   // over E*NHEAD
    if (i >= E * NHEAD) return;
    const int e = i >> 2, h = i & 3;
    const int s = src[e], d = dst[e];
    float v = el[s * NHEAD + h] + er[d * NHEAD + h];
    v = v > 0.f ? v : v * NEG_SLOPE;
    e_out[i] = v;
    rg_atomic_max_f32(&m[d * NHEAD + h], v);
}

// pass 2: ex = exp(e - m[dst]); segment sum into s[dst]
__global__ void rg_edge_expsum(const int* __restrict__ dst,
                               float* __restrict__ e_inout,
                               const float* __restrict__ m,
                               float* __restrict__ ssum, int E) {
    int i = blockIdx.x * blockDim.x + threadIdx.x;
    if (i >= E * NHEAD) return;
    const int e = i >> 2, h = i & 3;
    const int d = dst[e];
    float v = __expf(e_inout[i] - m[d * NHEAD + h]);
    e_inout[i] = v;
    atomicAdd(&ssum[d * NHEAD + h], v);
}

// pass 3: h2[dst] += (ex/s[dst]) * fs[src]
__global__ void rg_edge_message(const int* __restrict__ src,
                                const int* __restrict__ dst,
                                const float* __restrict__ ex,
                                const float* __restrict__ ssum,
                                const float* __restrict__ fs,
                                float* __restrict__ h2, int E) {
    int i = blockIdx.x * blockDim.x + threadIdx.x;   // over E*NHEAD
    if (i >= E * NHEAD) return;
    const int e = i >> 2, h = i & 3;
    const int s = src[e], d = dst[e];
    const float alpha = ex[i] / ssum[d * NHEAD + h];
    const float* fp = fs + (long)s * FEAT + h * DHEAD;
    float* op = h2 + (long)d * FEAT + h * DHEAD;
    __builtin_prefetch(fp, 0, 0);                    // global_prefetch_b8 on the gather
#pragma unroll
    for (int dd = 0; dd < DHEAD; ++dd)
        atomicAdd(&op[dd], alpha * fp[dd]);
}

// h2[n,c] += bias[c]
__global__ void rg_add_bias(float* __restrict__ h2, const float* __restrict__ bias, long n) {
    long i = (long)blockIdx.x * blockDim.x + threadIdx.x;
    if (i >= n) return;
    h2[i] += bias[i & (FEAT - 1)];
}

// hx = h1 + relu(h2)
__global__ void rg_add_relu(const float* __restrict__ h1, const float* __restrict__ h2,
                            float* __restrict__ hx, long n) {
    long i = (long)blockIdx.x * blockDim.x + threadIdx.x;
    if (i >= n) return;
    hx[i] = h1[i] + fmaxf(h2[i], 0.f);
}

// batchnorm stats: per-channel sum & sumsq (C=128, blockDim=256 -> 2 rows/block)
__global__ void rg_bn_stats(const float* __restrict__ x, float* __restrict__ sums,
                            float* __restrict__ sumsq, int n) {
    const int c = threadIdx.x & (FEAT - 1);
    const int rowsPerBlock = blockDim.x / FEAT;
    int r = blockIdx.x * rowsPerBlock + threadIdx.x / FEAT;
    const int stride = gridDim.x * rowsPerBlock;
    float s = 0.f, s2 = 0.f;
    for (; r < n; r += stride) {
        float v = x[(long)r * FEAT + c];
        s += v; s2 += v * v;
    }
    atomicAdd(&sums[c], s);
    atomicAdd(&sumsq[c], s2);
}

// y = g*(x-mu)*rsqrt(var+eps)+b  (optional fused relu)
__global__ void rg_bn_apply(const float* __restrict__ x, const float* __restrict__ sums,
                            const float* __restrict__ sumsq, const float* __restrict__ g,
                            const float* __restrict__ b, float* __restrict__ y,
                            int n, int doRelu) {
    long i = (long)blockIdx.x * blockDim.x + threadIdx.x;
    if (i >= (long)n * FEAT) return;
    const int c = (int)(i & (FEAT - 1));
    const float mu = sums[c] / n;
    const float var = sumsq[c] / n - mu * mu;
    float v = g[c] * (x[i] - mu) * rsqrtf(var + BN_EPS) + b[c];
    if (doRelu) v = fmaxf(v, 0.f);
    y[i] = v;
}

// ---------------------------------------------------------------------------
extern "C" void kernel_launch(void* const* d_in, const int* in_sizes, int n_in,
                              void* d_out, int out_size, void* d_ws, size_t ws_size,
                              hipStream_t stream) {
    const float* feat      = (const float*)d_in[0];
    const float* fc_W      = (const float*)d_in[1];   // [2,2,128,128]
    const float* attn_l    = (const float*)d_in[2];   // [2,2,4,32]
    const float* attn_r    = (const float*)d_in[3];
    const float* conv_bias = (const float*)d_in[4];
    const float* skip_W    = (const float*)d_in[5];   // [2,128,128]
    const float* skip_b    = (const float*)d_in[6];
    const float* norm_g    = (const float*)d_in[7];
    const float* norm_b    = (const float*)d_in[8];
    const float* clf_W1    = (const float*)d_in[9];
    const float* clf_b1    = (const float*)d_in[10];
    const float* clf_g     = (const float*)d_in[11];
    const float* clf_b     = (const float*)d_in[12];
    const float* clf_W2    = (const float*)d_in[13];  // [128,16]
    const float* clf_b2    = (const float*)d_in[14];
    const int*   src_all   = (const int*)d_in[15];    // [2,E]
    const int*   dst_all   = (const int*)d_in[16];    // [2,E]
    float* out = (float*)d_out;                       // [N,16]

    const long NF = (long)N_NODES * FEAT;
    float* w = (float*)d_ws;
    long o = 0;
    float* hA   = w + o; o += NF;
    float* hB   = w + o; o += NF;
    float* h1   = w + o; o += NF;
    float* h2   = w + o; o += NF;
    float* fs   = w + o; o += NF;
    float* el   = w + o; o += (long)N_NODES * NHEAD;
    float* er   = w + o; o += (long)N_NODES * NHEAD;
    float* mseg = w + o; o += (long)N_NODES * NHEAD;
    float* sseg = w + o; o += (long)N_NODES * NHEAD;
    float* e_s  = w + o; o += (long)N_EDGES * NHEAD;
    float* sums = w + o; o += FEAT;
    float* sumsq= w + o; o += FEAT;

    const int T = 256;
    auto blocks = [](long n) { return (int)((n + 255) / 256); };

    // 16x32 strip per wave for wide GEMMs, 16x16 for the 16-col classifier GEMM
    auto gemm2 = [&](const float* A, const float* B, const float* bias, float* Y,
                     int M, int K, int Nc) {
        const int nTiles = (M / 16) * (Nc / 32);
        const int nBlocks = (nTiles + 7) / 8;                 // 8 waves / 256-thread block
        rg_wmma_gemm<2><<<nBlocks, 256, 0, stream>>>(A, B, bias, Y, M, K, Nc);
    };
    auto gemm1 = [&](const float* A, const float* B, const float* bias, float* Y,
                     int M, int K, int Nc) {
        const int nTiles = (M / 16) * (Nc / 16);
        const int nBlocks = (nTiles + 7) / 8;
        rg_wmma_gemm<1><<<nBlocks, 256, 0, stream>>>(A, B, bias, Y, M, K, Nc);
    };
    auto bn = [&](const float* x, const float* g, const float* b, float* y, int relu) {
        rg_fill<<<1, 256, 0, stream>>>(sums, 0.f, FEAT);
        rg_fill<<<1, 256, 0, stream>>>(sumsq, 0.f, FEAT);
        rg_bn_stats<<<512, 256, 0, stream>>>(x, sums, sumsq, N_NODES);
        rg_bn_apply<<<blocks(NF), T, 0, stream>>>(x, sums, sumsq, g, b, y, N_NODES, relu);
    };

    const long EH = (long)N_EDGES * NHEAD;
    const long NH = (long)N_NODES * NHEAD;
    const float* cur = feat;

    for (int l = 0; l < 2; ++l) {
        // skip linear
        gemm2(cur, skip_W + (long)l * FEAT * FEAT, skip_b + l * FEAT, h1, N_NODES, FEAT, FEAT);
        rg_fill<<<blocks(NF), T, 0, stream>>>(h2, 0.f, NF);

        for (int r = 0; r < 2; ++r) {
            const long pr = (long)(l * 2 + r);
            const int* src = src_all + (long)r * N_EDGES;
            const int* dst = dst_all + (long)r * N_EDGES;

            gemm2(cur, fc_W + pr * FEAT * FEAT, nullptr, fs, N_NODES, FEAT, FEAT);
            rg_node_attn<<<blocks(NH), T, 0, stream>>>(fs, attn_l + pr * FEAT,
                                                       attn_r + pr * FEAT, el, er, N_NODES);
            rg_fill<<<blocks(NH), T, 0, stream>>>(mseg, -3.402823466e+38f, NH);
            rg_fill<<<blocks(NH), T, 0, stream>>>(sseg, 0.f, NH);
            rg_edge_logits_max<<<blocks(EH), T, 0, stream>>>(el, er, src, dst, e_s, mseg, N_EDGES);
            rg_edge_expsum<<<blocks(EH), T, 0, stream>>>(dst, e_s, mseg, sseg, N_EDGES);
            rg_edge_message<<<blocks(EH), T, 0, stream>>>(src, dst, e_s, sseg, fs, h2, N_EDGES);
            rg_add_bias<<<blocks(NF), T, 0, stream>>>(h2, conv_bias + pr * FEAT, NF);
        }
        // hx = h1 + relu(h2)  (reuse fs as hx)
        rg_add_relu<<<blocks(NF), T, 0, stream>>>(h1, h2, fs, NF);
        float* hn = (l == 0) ? hA : hB;
        bn(fs, norm_g + l * FEAT, norm_b + l * FEAT, hn, 0);
        cur = hn;
    }

    // classifier: bn(relu fused) then final 128x16 GEMM into d_out
    gemm2(cur, clf_W1, clf_b1, h1, N_NODES, FEAT, FEAT);
    bn(h1, clf_g, clf_b, h2, 1);
    gemm1(h2, clf_W2, clf_b2, out, N_NODES, FEAT, 16);
}
